// Classify_node_128849019549
// MI455X (gfx1250) — matrix operand order repaired and checked by reading, not compiled, not run on hardware
//
#include <hip/hip_runtime.h>
#include <hip/hip_bf16.h>
#include <math.h>

typedef __attribute__((ext_vector_type(16))) _Float16 v16h;
typedef __attribute__((ext_vector_type(8)))  _Float16 v8h;
typedef __attribute__((ext_vector_type(8)))  float    v8f;

#define FDIM 128   // feature / hidden dim
#define CPAD 48    // padded classifier cols (C=40 -> 3 tiles of 16)
#define NCLS 40

// ---------------- degree / norm ----------------
__global__ void k_deg_init(float* deg, int n) {
    int i = blockIdx.x * blockDim.x + threadIdx.x;
    if (i < n) deg[i] = 1.0f;                 // self loop
}
__global__ void k_deg_count(const int* __restrict__ dst, float* deg, int e) {
    int i = blockIdx.x * blockDim.x + threadIdx.x;
    if (i < e) unsafeAtomicAdd(&deg[dst[i]], 1.0f);
}
__global__ void k_deg_finish(float* deg, int n) {
    int i = blockIdx.x * blockDim.x + threadIdx.x;
    if (i < n) deg[i] = rsqrtf(deg[i]);       // deg >= 1 always
}

// ---------------- conversions ----------------
__global__ void k_f32_to_f16(const float* __restrict__ src, _Float16* __restrict__ dst, int n) {
    int i = blockIdx.x * blockDim.x + threadIdx.x;
    if (i < n) dst[i] = (_Float16)src[i];
}
// W [K=128][N=128] f32 row-major -> WT [n][k] f16
__global__ void k_w_transpose(const float* __restrict__ W, _Float16* __restrict__ WT) {
    int i = blockIdx.x * blockDim.x + threadIdx.x;   // 128*128 threads
    int n = i >> 7, k = i & 127;
    WT[i] = (_Float16)W[k * FDIM + n];
}
// Wl [128][40] -> WlT [48][128] f16, zero pad cols >= 40
__global__ void k_wl_transpose(const float* __restrict__ Wl, _Float16* __restrict__ WT) {
    int i = blockIdx.x * blockDim.x + threadIdx.x;   // 48*128 threads
    if (i >= CPAD * FDIM) return;
    int c = i >> 7, k = i & 127;
    WT[i] = (c < NCLS) ? (_Float16)Wl[k * NCLS + c] : (_Float16)0.0f;
}
__global__ void k_zero_f4(float4* p, int n4) {
    int i = blockIdx.x * blockDim.x + threadIdx.x;
    if (i < n4) p[i] = make_float4(0.f, 0.f, 0.f, 0.f);
}

// ---------------- WMMA GEMM:  C[rowTiles*16, NCT*16] = A[., 128] x BT[NCT*16][128]^T -------
// One wave computes a full 16-row x (NCT*16)-col slab: A loaded once per k-step,
// NCT independent accumulators -> NCT back-to-back independent WMMAs per k-step
// (hides the 5-slot WMMA->WMMA RAW hazard) and 8x less A traffic than tile-per-wave.
template <int NCT>
__global__ void k_gemm_wmma(const _Float16* __restrict__ A,
                            const _Float16* __restrict__ BT,
                            float* __restrict__ C, int ldc, int rowTiles) {
    const int lane = threadIdx.x & 31;
    const int wave = threadIdx.x >> 5;
    const int rt   = blockIdx.x * (blockDim.x >> 5) + wave;   // row tile
    if (rt >= rowTiles) return;                                // uniform per wave
    const int l15 = lane & 15;
    const int kh  = lane >> 4;                                 // lane half: 0/1
    const int row = rt * 16 + l15;

    const _Float16* Arow  = A  + (size_t)row * FDIM;
    const _Float16* Bbase = BT + (size_t)l15 * FDIM;           // + ct*16*FDIM per col tile

    v8f acc[NCT];
#pragma unroll
    for (int ct = 0; ct < NCT; ++ct) acc[ct] = v8f{};

#pragma unroll
    for (int kt = 0; kt < 4; ++kt) {
        const int k0 = kt * 32;
        // A 16x32: elems 0..7 -> K = k0+kh*8+i ; elems 8..15 -> K = k0+16+kh*8+i
        v8h a0 = *(const v8h*)(Arow + k0 + kh * 8);
        v8h a1 = *(const v8h*)(Arow + k0 + 16 + kh * 8);
        v16h a;
#pragma unroll
        for (int i = 0; i < 8; ++i) { a[i] = a0[i]; a[i + 8] = a1[i]; }
#pragma unroll
        for (int ct = 0; ct < NCT; ++ct) {
            // B 32x16: lane holds column ct*16 + l15, elems 0..7 -> K = k0+kh*8+i ; elems 8..15 -> K = k0+16+kh*8+i
            const _Float16* Bcol = Bbase + (size_t)ct * 16 * FDIM + k0 + kh * 8;
            v8h b0 = *(const v8h*)(Bcol);
            v8h b1 = *(const v8h*)(Bcol + 16);
            v16h b;
#pragma unroll
            for (int i = 0; i < 8; ++i) { b[i] = b0[i]; b[i + 8] = b1[i]; }
            acc[ct] = __builtin_amdgcn_wmma_f32_16x16x32_f16(
                false, a, false, b, (short)0, acc[ct], false, false);
        }
    }
    // D layout: VGPR r holds M = r + 8*(lane>=16), N = lane%16
    const int mbase = kh * 8;
#pragma unroll
    for (int ct = 0; ct < NCT; ++ct) {
#pragma unroll
        for (int r = 0; r < 8; ++r)
            C[(size_t)(rt * 16 + mbase + r) * ldc + ct * 16 + l15] = acc[ct][r];
    }
}

// ---------------- edge scatter:  agg[dst] += h[src] * dinv[src]*dinv[dst] ----------------
// one wave per edge, lane covers 4 floats (float4); atomics hit the 192MB L2
__global__ void k_scatter(const int* __restrict__ src, const int* __restrict__ dst,
                          const float* __restrict__ dinv,
                          const float* __restrict__ h,
                          float* __restrict__ agg, int e) {
    const int edge = blockIdx.x * 8 + (threadIdx.x >> 5);
    if (edge >= e) return;
    const int lane = threadIdx.x & 31;
    const int s = src[edge], d = dst[edge];
    const float nrm = dinv[s] * dinv[d];
    const float4 v = *(const float4*)(h + (size_t)s * FDIM + lane * 4);
    float* out = agg + (size_t)d * FDIM + lane * 4;
    unsafeAtomicAdd(out + 0, v.x * nrm);
    unsafeAtomicAdd(out + 1, v.y * nrm);
    unsafeAtomicAdd(out + 2, v.z * nrm);
    unsafeAtomicAdd(out + 3, v.w * nrm);
}

// ---------------- epilogue: self-loop + bias + ELU, write f16 for next GEMM ----------------
__global__ void k_epilogue(const float* __restrict__ agg, const float* __restrict__ hlin,
                           const float* __restrict__ dinv, const float* __restrict__ bias,
                           _Float16* __restrict__ h16, int total) {
    int i = blockIdx.x * blockDim.x + threadIdx.x;
    if (i >= total) return;
    const int node = i >> 7, f = i & 127;
    const float di = dinv[node];
    const float v = agg[i] + hlin[i] * di * di + bias[f];
    const float e = v > 0.f ? v : expm1f(v);      // ELU, alpha=1
    h16[i] = (_Float16)e;
}

// ---------------- log-softmax over 40 classes, one wave per node ----------------
__global__ void k_logsoftmax(const float* __restrict__ logits, const float* __restrict__ bl,
                             float* __restrict__ out, int n) {
    const int node = blockIdx.x * 8 + (threadIdx.x >> 5);
    if (node >= n) return;
    const int lane = threadIdx.x & 31;
    const float* L = logits + (size_t)node * CPAD;
    const float v0 = L[lane] + bl[lane];                               // lane < 32 < 40: valid
    const float v1 = (lane < NCLS - 32) ? L[lane + 32] + bl[lane + 32] : -1e30f;
    float m = fmaxf(v0, v1);
#pragma unroll
    for (int off = 16; off >= 1; off >>= 1) m = fmaxf(m, __shfl_xor(m, off, 32));
    float s = __expf(v0 - m) + ((lane < NCLS - 32) ? __expf(v1 - m) : 0.f);
#pragma unroll
    for (int off = 16; off >= 1; off >>= 1) s += __shfl_xor(s, off, 32);
    const float lse = m + __logf(s);
    float* O = out + (size_t)node * NCLS;
    O[lane] = v0 - lse;
    if (lane < NCLS - 32) O[lane + 32] = v1 - lse;
}

static inline size_t align256(size_t x) { return (x + 255) & ~(size_t)255; }

extern "C" void kernel_launch(void* const* d_in, const int* in_sizes, int n_in,
                              void* d_out, int out_size, void* d_ws, size_t ws_size,
                              hipStream_t stream) {
    const float* x   = (const float*)d_in[0];
    const int*   ei  = (const int*)d_in[1];
    const float* W0  = (const float*)d_in[2];
    const float* b0  = (const float*)d_in[3];
    const float* W1  = (const float*)d_in[4];
    const float* b1  = (const float*)d_in[5];
    const float* Wl  = (const float*)d_in[6];
    const float* bl  = (const float*)d_in[7];
    float* out = (float*)d_out;

    const int N = in_sizes[0] / FDIM;       // 100000
    const int E = in_sizes[1] / 2;          // 1600000
    const int* src = ei;
    const int* dst = ei + E;

    // workspace carve-up
    char* ws = (char*)d_ws;
    size_t off = 0;
    float*    dinv   = (float*)(ws + off);    off += align256((size_t)N * 4);
    float*    hlin   = (float*)(ws + off);    off += align256((size_t)N * FDIM * 4);
    float*    agg    = (float*)(ws + off);    off += align256((size_t)N * FDIM * 4);
    _Float16* h16    = (_Float16*)(ws + off); off += align256((size_t)N * FDIM * 2);
    _Float16* wt0    = (_Float16*)(ws + off); off += align256((size_t)FDIM * FDIM * 2);
    _Float16* wt1    = (_Float16*)(ws + off); off += align256((size_t)FDIM * FDIM * 2);
    _Float16* wlt    = (_Float16*)(ws + off); off += align256((size_t)CPAD * FDIM * 2);
    float*    logits = (float*)(ws + off);    off += align256((size_t)N * CPAD * 4);

    const int NF       = N * FDIM;
    const int gNF      = (NF + 255) / 256;
    const int gN       = (N + 255) / 256;
    const int gE       = (E + 255) / 256;
    const int rowTiles = (N + 15) / 16;          // 6250
    const int gGemm    = (rowTiles + 3) / 4;     // 4 waves (128 thr) per block
    const int gEdgeW   = (E + 7) / 8;            // wave-per-edge blocks
    const int gNodeW   = (N + 7) / 8;            // wave-per-node blocks

    // degrees -> dinv
    k_deg_init  <<<gN, 256, 0, stream>>>(dinv, N);
    k_deg_count <<<gE, 256, 0, stream>>>(dst, dinv, E);
    k_deg_finish<<<gN, 256, 0, stream>>>(dinv, N);

    // weight prep
    k_w_transpose <<<(FDIM * FDIM) / 256, 256, 0, stream>>>(W0, wt0);
    k_w_transpose <<<(FDIM * FDIM) / 256, 256, 0, stream>>>(W1, wt1);
    k_wl_transpose<<<(CPAD * FDIM + 255) / 256, 256, 0, stream>>>(Wl, wlt);

    // ---- layer 0 ----
    k_f32_to_f16<<<gNF, 256, 0, stream>>>(x, h16, NF);
    k_gemm_wmma<8><<<gGemm, 128, 0, stream>>>(h16, wt0, hlin, FDIM, rowTiles);
    k_zero_f4   <<<(NF / 4 + 255) / 256, 256, 0, stream>>>((float4*)agg, NF / 4);
    k_scatter   <<<gEdgeW, 256, 0, stream>>>(src, dst, dinv, hlin, agg, E);
    k_epilogue  <<<gNF, 256, 0, stream>>>(agg, hlin, dinv, b0, h16, NF);

    // ---- layer 1 ----
    k_gemm_wmma<8><<<gGemm, 128, 0, stream>>>(h16, wt1, hlin, FDIM, rowTiles);
    k_zero_f4   <<<(NF / 4 + 255) / 256, 256, 0, stream>>>((float4*)agg, NF / 4);
    k_scatter   <<<gEdgeW, 256, 0, stream>>>(src, dst, dinv, hlin, agg, E);
    k_epilogue  <<<gNF, 256, 0, stream>>>(agg, hlin, dinv, b1, h16, NF);

    // ---- classifier (3 col tiles) + log-softmax ----
    k_gemm_wmma<3><<<gGemm, 128, 0, stream>>>(h16, wlt, logits, CPAD, rowTiles);
    k_logsoftmax<<<gNodeW, 256, 0, stream>>>(logits, bl, out, N);
}
